// MultiHeadAttention_31233002177208
// MI455X (gfx1250) — matrix operand order repaired and checked by reading, not compiled, not run on hardware
//
#include <hip/hip_runtime.h>
#include <hip/hip_bf16.h>

#define DM   1024
#define NH   16
#define HDM  64
#define BB   2
#define TT   2048
#define MROWS (BB * TT)   // 4096

typedef _Float16 f16;
typedef _Float16 v16h __attribute__((ext_vector_type(16)));
typedef _Float16 v8h  __attribute__((ext_vector_type(8)));
typedef _Float16 v4h  __attribute__((ext_vector_type(4)));
typedef float    v8f  __attribute__((ext_vector_type(8)));
typedef float    v4f  __attribute__((ext_vector_type(4)));

static __device__ __forceinline__ v16h concat8(v8h lo, v8h hi) {
  return __builtin_shufflevector(lo, hi, 0,1,2,3,4,5,6,7,8,9,10,11,12,13,14,15);
}

// A-matrix fragment (16x32 f16, MxK): lane holds row m=lane&15; half=lane>>4.
// slot j -> k = (j/8)*16 + half*8 + (j%8)  => two 8-half runs at half*8 and 16+half*8.
static __device__ __forceinline__ v16h load_a_frag(const f16* rowBase, int hf) {
  v8h lo = *(const v8h*)(rowBase + hf * 8);
  v8h hi = *(const v8h*)(rowBase + 16 + hf * 8);
  return concat8(lo, hi);
}

// B-matrix fragment (32x16 f16, KxN): lane holds col n=lane&15; half=lane>>4.
// slot j -> k = (j/8)*16 + half*8 + (j%8)  => two 8-half runs at half*8 and 16+half*8.
static __device__ __forceinline__ v16h load_b_frag(const f16* rowBase, int hf) {
  v8h lo = *(const v8h*)(rowBase + hf * 8);
  v8h hi = *(const v8h*)(rowBase + 16 + hf * 8);
  return concat8(lo, hi);
}

static __device__ __forceinline__ v8f wmma_f16(v16h a, v16h b, v8f c) {
  return __builtin_amdgcn_wmma_f32_16x16x32_f16(false, a, false, b, (short)0, c, false, false);
}

// async global->LDS 16B copy per lane (ASYNCcnt-tracked), CDNA5 path
static __device__ __forceinline__ void async_copy_b128(void* lds_dst, const void* gsrc) {
  unsigned loff = (unsigned)(uintptr_t)lds_dst;                 // addr[31:0] = LDS offset
  unsigned long long ga = (unsigned long long)(uintptr_t)gsrc;
  asm volatile("global_load_async_to_lds_b128 %0, %1, off" :: "v"(loff), "v"(ga) : "memory");
}

static __device__ __forceinline__ v4h cvt4(v4f v) {
  v4h r = {(f16)v[0], (f16)v[1], (f16)v[2], (f16)v[3]};
  return r;
}

#define XS_STRIDE 40   // 32 + 8 pad halves (80B rows -> conflict-free 16-row b128)

// ---------------------------------------------------------------------------
// Kernel 1: Q/K/V projections  Y = X * W^T  (M=4096, N=1024, K=1024)
// grid (M/128, N/64, 3), block 128 (4 waves). Wave: 32-row strip x 64 cols.
// Register-prefetch pipelined over K tiles.
// ---------------------------------------------------------------------------
__global__ __launch_bounds__(128) void qkv_proj_kernel(
    const float* __restrict__ x,  const float* __restrict__ Wq,
    const float* __restrict__ Wk, const float* __restrict__ Wv,
    f16* __restrict__ q_ws, f16* __restrict__ k_ws, f16* __restrict__ vT_ws)
{
  __shared__ f16 Xs[128 * XS_STRIDE];
  __shared__ f16 Ws[64 * XS_STRIDE];
  const int tid  = threadIdx.x;
  const int wave = tid >> 5, lane = tid & 31;
  const int lm   = lane & 15, hf = lane >> 4;
  const int m0 = blockIdx.x * 128;
  const int n0 = blockIdx.y * 64;
  const int which = blockIdx.z;
  const float* W = (which == 0) ? Wq : (which == 1) ? Wk : Wv;

  const v8f vzero = {0.f,0.f,0.f,0.f,0.f,0.f,0.f,0.f};
  v8f acc[2][4] = {{vzero, vzero, vzero, vzero}, {vzero, vzero, vzero, vzero}};

  // per-thread staging coordinates
  int xrow[8], xc[8], wrow[4], wc[4];
  #pragma unroll
  for (int i = 0; i < 8; ++i) { int f = tid + i * 128; xrow[i] = f >> 3; xc[i] = (f & 7) * 4; }
  #pragma unroll
  for (int i = 0; i < 4; ++i) { int f = tid + i * 128; wrow[i] = f >> 3; wc[i] = (f & 7) * 4; }

  // prologue: fetch first K-tile into registers
  v4f xr[8], wr[4];
  #pragma unroll
  for (int i = 0; i < 8; ++i)
    xr[i] = *(const v4f*)(x + (size_t)(m0 + xrow[i]) * DM + xc[i]);
  #pragma unroll
  for (int i = 0; i < 4; ++i)
    wr[i] = *(const v4f*)(W + (size_t)(n0 + wrow[i]) * DM + wc[i]);

  for (int kk = 0; kk < DM; kk += 32) {
    #pragma unroll
    for (int i = 0; i < 8; ++i) *(v4h*)&Xs[xrow[i] * XS_STRIDE + xc[i]] = cvt4(xr[i]);
    #pragma unroll
    for (int i = 0; i < 4; ++i) *(v4h*)&Ws[wrow[i] * XS_STRIDE + wc[i]] = cvt4(wr[i]);
    __syncthreads();

    if (kk + 32 < DM) {   // prefetch next K-tile while WMMAs run
      #pragma unroll
      for (int i = 0; i < 8; ++i)
        xr[i] = *(const v4f*)(x + (size_t)(m0 + xrow[i]) * DM + kk + 32 + xc[i]);
      #pragma unroll
      for (int i = 0; i < 4; ++i)
        wr[i] = *(const v4f*)(W + (size_t)(n0 + wrow[i]) * DM + kk + 32 + wc[i]);
    }

    v16h a0 = load_a_frag(&Xs[(wave * 32 +      lm) * XS_STRIDE], hf);
    v16h a1 = load_a_frag(&Xs[(wave * 32 + 16 + lm) * XS_STRIDE], hf);
    #pragma unroll
    for (int nt = 0; nt < 4; ++nt) {
      v16h b = load_b_frag(&Ws[(nt * 16 + lm) * XS_STRIDE], hf);
      acc[0][nt] = wmma_f16(a0, b, acc[0][nt]);
      acc[1][nt] = wmma_f16(a1, b, acc[1][nt]);
    }
    __syncthreads();
  }

  // affine store: b and h are block-uniform (128 | 2048, 64-col tile == 1 head)
  const int bidx = m0 >> 11;
  const int h    = n0 >> 6;
  const int tb   = (m0 & (TT - 1)) + wave * 32;
  if (which != 2) {
    f16* dst = ((which == 0) ? q_ws : k_ws) + (((size_t)bidx * NH + h) * TT) * HDM;
    #pragma unroll
    for (int st = 0; st < 2; ++st)
      #pragma unroll
      for (int nt = 0; nt < 4; ++nt)
        #pragma unroll
        for (int r = 0; r < 8; ++r) {
          int t = tb + st * 16 + r + 8 * hf;
          int d = nt * 16 + lm;
          dst[(size_t)t * HDM + d] = (f16)acc[st][nt][r];
        }
  } else {
    // transposed V: lane's 8 accum rows are consecutive t -> packed b128 store
    f16* dst = vT_ws + (((size_t)bidx * NH + h) * HDM) * TT;
    #pragma unroll
    for (int st = 0; st < 2; ++st)
      #pragma unroll
      for (int nt = 0; nt < 4; ++nt) {
        int t0 = tb + st * 16 + 8 * hf;
        int d  = nt * 16 + lm;
        v8h pk = {(f16)acc[st][nt][0], (f16)acc[st][nt][1],
                  (f16)acc[st][nt][2], (f16)acc[st][nt][3],
                  (f16)acc[st][nt][4], (f16)acc[st][nt][5],
                  (f16)acc[st][nt][6], (f16)acc[st][nt][7]};
        *(v8h*)&dst[(size_t)d * TT + t0] = pk;
      }
  }
}

// ---------------------------------------------------------------------------
// Kernel 2: causal flash attention. grid (T/64, B*H), block 128 (4 waves).
// Wave w owns queries [q0+16w, q0+16w+16); streams keys in chunks of 32.
// K/V chunks double-buffered via GLOBAL_LOAD_ASYNC_TO_LDS_B128 (ASYNCcnt).
// ---------------------------------------------------------------------------
#define KS_STRIDE 72
#define VS_STRIDE 40
#define PS_STRIDE 40

__global__ __launch_bounds__(128) void attn_kernel(
    const f16* __restrict__ q_ws, const f16* __restrict__ k_ws,
    const f16* __restrict__ vT_ws, f16* __restrict__ o_ws)
{
  __shared__ f16 Ks[2][32 * KS_STRIDE];     // [buf][key s][d]
  __shared__ f16 Vs[2][64 * VS_STRIDE];     // [buf][d][key s]  (from V^T)
  __shared__ f16 Ps[4 * 16 * PS_STRIDE];    // per-wave P tile [q][s]
  const int tid  = threadIdx.x;
  const int wave = tid >> 5, lane = tid & 31;
  const int lm   = lane & 15, hf = lane >> 4;
  const int q0 = blockIdx.x * 64;
  const int bh = blockIdx.y;
  const int qw = q0 + wave * 16;

  const f16* qp = q_ws  + (size_t)bh * TT * HDM;
  const f16* kp = k_ws  + (size_t)bh * TT * HDM;
  const f16* vp = vT_ws + (size_t)bh * HDM * TT;

  // preload Q fragments for this wave's 16-row strip (d 0..31, 32..63)
  const f16* qrow = qp + (size_t)(qw + lm) * HDM;
  v16h aq0 = load_a_frag(qrow, hf);
  v16h aq1 = load_a_frag(qrow + 32, hf);

  const v8f vzero = {0.f,0.f,0.f,0.f,0.f,0.f,0.f,0.f};
  v8f oa[4] = {vzero, vzero, vzero, vzero};
  float row_m[8], row_l[8];
  #pragma unroll
  for (int r = 0; r < 8; ++r) { row_m[r] = -1e30f; row_l[r] = 0.f; }

  f16* Pw = &Ps[wave * 16 * PS_STRIDE];
  // per-thread staging coordinates (hoisted)
  const int f0 = tid,       kr0 = f0 >> 3, kc0 = (f0 & 7) * 8, vr0 = f0 >> 2, vc0 = (f0 & 3) * 8;
  const int f1 = tid + 128, kr1 = f1 >> 3, kc1 = (f1 & 7) * 8, vr1 = f1 >> 2, vc1 = (f1 & 3) * 8;

  const int end = q0 + 64;

  // issue chunk 0 into buffer 0 (4 async ops per wave per chunk)
  async_copy_b128(&Ks[0][kr0 * KS_STRIDE + kc0], kp + (size_t)kr0 * HDM + kc0);
  async_copy_b128(&Ks[0][kr1 * KS_STRIDE + kc1], kp + (size_t)kr1 * HDM + kc1);
  async_copy_b128(&Vs[0][vr0 * VS_STRIDE + vc0], vp + (size_t)vr0 * TT + vc0);
  async_copy_b128(&Vs[0][vr1 * VS_STRIDE + vc1], vp + (size_t)vr1 * TT + vc1);

  for (int s0 = 0; s0 < end; s0 += 32) {
    const int  buf  = (s0 >> 5) & 1;
    const bool more = (s0 + 32) < end;
    if (more) {   // prefetch next chunk into other buffer (WAR safe: see barrier below)
      const int sn = s0 + 32, nb = buf ^ 1;
      async_copy_b128(&Ks[nb][kr0 * KS_STRIDE + kc0], kp + (size_t)(sn + kr0) * HDM + kc0);
      async_copy_b128(&Ks[nb][kr1 * KS_STRIDE + kc1], kp + (size_t)(sn + kr1) * HDM + kc1);
      async_copy_b128(&Vs[nb][vr0 * VS_STRIDE + vc0], vp + (size_t)vr0 * TT + sn + vc0);
      async_copy_b128(&Vs[nb][vr1 * VS_STRIDE + vc1], vp + (size_t)vr1 * TT + sn + vc1);
      asm volatile("s_wait_asynccnt 4" ::: "memory");   // oldest 4 (this chunk) done
    } else {
      asm volatile("s_wait_asynccnt 0" ::: "memory");
    }
    __syncthreads();   // all waves' staging of this chunk complete

    if (s0 <= qw + 15) {   // wave-uniform: chunk intersects causal window
      float p0[8], p1[8];
      #pragma unroll
      for (int kt = 0; kt < 2; ++kt) {
        const f16* krow0 = &Ks[buf][(kt * 16 + lm) * KS_STRIDE];
        v16h bk0 = load_b_frag(krow0, hf);        // d 0..31
        v16h bk1 = load_b_frag(krow0 + 32, hf);   // d 32..63
        v8f s = vzero;
        s = wmma_f16(aq0, bk0, s);
        s = wmma_f16(aq1, bk1, s);
        float* pp = kt ? p1 : p0;
        #pragma unroll
        for (int r = 0; r < 8; ++r) {
          int tq = qw + r + 8 * hf;
          int sg = s0 + kt * 16 + lm;
          float v = s[r] * 0.125f;                // 1/sqrt(64)
          if (sg > tq) v = -1e30f;                // causal mask
          pp[r] = v;
        }
      }
      // online softmax per row (16 lanes of each half hold one row's keys)
      #pragma unroll
      for (int r = 0; r < 8; ++r) {
        float mx = fmaxf(p0[r], p1[r]);
        mx = fmaxf(mx, __shfl_xor(mx, 1, 32));
        mx = fmaxf(mx, __shfl_xor(mx, 2, 32));
        mx = fmaxf(mx, __shfl_xor(mx, 4, 32));
        mx = fmaxf(mx, __shfl_xor(mx, 8, 32));
        float nm   = fmaxf(row_m[r], mx);
        float corr = __expf(row_m[r] - nm);
        row_m[r] = nm;
        p0[r] = __expf(p0[r] - nm);
        p1[r] = __expf(p1[r] - nm);
        float rs = p0[r] + p1[r];
        rs += __shfl_xor(rs, 1, 32);
        rs += __shfl_xor(rs, 2, 32);
        rs += __shfl_xor(rs, 4, 32);
        rs += __shfl_xor(rs, 8, 32);
        row_l[r] = row_l[r] * corr + rs;
        #pragma unroll
        for (int dt = 0; dt < 4; ++dt) oa[dt][r] = oa[dt][r] * corr;
        // stage P (C-layout -> row-major LDS tile)
        Pw[(r + 8 * hf) * PS_STRIDE + lm]      = (f16)p0[r];
        Pw[(r + 8 * hf) * PS_STRIDE + 16 + lm] = (f16)p1[r];
      }
      asm volatile("s_wait_dscnt 0" ::: "memory");  // wave-local LDS RAW
      v16h ap = load_a_frag(&Pw[lm * PS_STRIDE], hf);
      #pragma unroll
      for (int dt = 0; dt < 4; ++dt) {
        v16h bv = load_b_frag(&Vs[buf][(dt * 16 + lm) * VS_STRIDE], hf);
        oa[dt] = wmma_f16(ap, bv, oa[dt]);
      }
    }
    __syncthreads();   // all reads of buf done before it is refilled next+1 iter
  }

  // epilogue: normalize and store to o_ws[B*T][D] (col = h*64 + d)
  const int b = bh >> 4, h = bh & 15;
  f16* orow = o_ws + (size_t)(b * TT) * DM + (size_t)h * HDM;
  #pragma unroll
  for (int dt = 0; dt < 4; ++dt) {
    #pragma unroll
    for (int r = 0; r < 8; ++r) {
      int tq = qw + r + 8 * hf;
      int d  = dt * 16 + lm;
      float val = oa[dt][r] / row_l[r];
      orow[(size_t)tq * DM + d] = (f16)val;
    }
  }
}

// ---------------------------------------------------------------------------
// Kernel 3: out = O * Wo^T + bo   (A already f16 in workspace, fp32 output)
// grid (M/128, N/64), block 128 (4 waves). Register-prefetch pipelined.
// ---------------------------------------------------------------------------
__global__ __launch_bounds__(128) void out_proj_kernel(
    const f16* __restrict__ o_ws, const float* __restrict__ Wo,
    const float* __restrict__ bo, float* __restrict__ out)
{
  __shared__ f16 Xs[128 * XS_STRIDE];
  __shared__ f16 Ws[64 * XS_STRIDE];
  const int tid  = threadIdx.x;
  const int wave = tid >> 5, lane = tid & 31;
  const int lm   = lane & 15, hf = lane >> 4;
  const int m0 = blockIdx.x * 128;
  const int n0 = blockIdx.y * 64;

  const v8f vzero = {0.f,0.f,0.f,0.f,0.f,0.f,0.f,0.f};
  v8f acc[2][4] = {{vzero, vzero, vzero, vzero}, {vzero, vzero, vzero, vzero}};

  int arow[4], ac[4], wrow[4], wc[4];
  #pragma unroll
  for (int i = 0; i < 4; ++i) {
    int f = tid + i * 128;
    arow[i] = f >> 2; ac[i] = (f & 3) * 8;     // A tile: 512 v8h chunks
    wrow[i] = f >> 3; wc[i] = (f & 7) * 4;     // W tile: 512 v4f chunks
  }

  v8h arx[4]; v4f wrx[4];
  #pragma unroll
  for (int i = 0; i < 4; ++i) {
    arx[i] = *(const v8h*)(o_ws + (size_t)(m0 + arow[i]) * DM + ac[i]);
    wrx[i] = *(const v4f*)(Wo + (size_t)(n0 + wrow[i]) * DM + wc[i]);
  }

  for (int kk = 0; kk < DM; kk += 32) {
    #pragma unroll
    for (int i = 0; i < 4; ++i) {
      *(v8h*)&Xs[arow[i] * XS_STRIDE + ac[i]] = arx[i];
      *(v4h*)&Ws[wrow[i] * XS_STRIDE + wc[i]] = cvt4(wrx[i]);
    }
    __syncthreads();

    if (kk + 32 < DM) {
      #pragma unroll
      for (int i = 0; i < 4; ++i) {
        arx[i] = *(const v8h*)(o_ws + (size_t)(m0 + arow[i]) * DM + kk + 32 + ac[i]);
        wrx[i] = *(const v4f*)(Wo + (size_t)(n0 + wrow[i]) * DM + kk + 32 + wc[i]);
      }
    }

    v16h a0 = load_a_frag(&Xs[(wave * 32 +      lm) * XS_STRIDE], hf);
    v16h a1 = load_a_frag(&Xs[(wave * 32 + 16 + lm) * XS_STRIDE], hf);
    #pragma unroll
    for (int nt = 0; nt < 4; ++nt) {
      v16h b = load_b_frag(&Ws[(nt * 16 + lm) * XS_STRIDE], hf);
      acc[0][nt] = wmma_f16(a0, b, acc[0][nt]);
      acc[1][nt] = wmma_f16(a1, b, acc[1][nt]);
    }
    __syncthreads();
  }

  #pragma unroll
  for (int nt = 0; nt < 4; ++nt) {
    const int C = n0 + nt * 16 + lm;
    const float bias = bo[C];
    #pragma unroll
    for (int st = 0; st < 2; ++st)
      #pragma unroll
      for (int r = 0; r < 8; ++r) {
        int R = m0 + wave * 32 + st * 16 + r + 8 * hf;
        out[(size_t)R * DM + C] = acc[st][nt][r] + bias;
      }
  }
}

// ---------------------------------------------------------------------------
extern "C" void kernel_launch(void* const* d_in, const int* in_sizes, int n_in,
                              void* d_out, int out_size, void* d_ws, size_t ws_size,
                              hipStream_t stream)
{
  const float* x  = (const float*)d_in[0];
  const float* Wq = (const float*)d_in[1];
  const float* Wk = (const float*)d_in[2];
  const float* Wv = (const float*)d_in[3];
  const float* Wo = (const float*)d_in[4];
  const float* bo = (const float*)d_in[5];
  float* out = (float*)d_out;

  f16* q_ws  = (f16*)d_ws;
  f16* k_ws  = q_ws  + (size_t)MROWS * DM;
  f16* vT_ws = k_ws  + (size_t)MROWS * DM;
  f16* o_ws  = vT_ws + (size_t)MROWS * DM;   // total 32 MB f16 workspace

  qkv_proj_kernel<<<dim3(MROWS / 128, DM / 64, 3), 128, 0, stream>>>(
      x, Wq, Wk, Wv, q_ws, k_ws, vT_ws);
  attn_kernel<<<dim3(TT / 64, BB * NH), 128, 0, stream>>>(
      q_ws, k_ws, vT_ws, o_ws);
  out_proj_kernel<<<dim3(MROWS / 128, DM / 64), 128, 0, stream>>>(
      o_ws, Wo, bo, out);
}
